// Model_15058155340510
// MI455X (gfx1250) — matrix operand order repaired and checked by reading, not compiled, not run on hardware
//
#include <hip/hip_runtime.h>
#include <hip/hip_bf16.h>

// ---------------- model constants ----------------
#define SL_C     4096
#define P_C      16
#define STRIDE_C 8
#define NP_C     511            // NPATCH
#define BM_C     64             // B*M sequences
#define ROWS_C   (BM_C * NP_C)  // 32704 (= 2044 * 16)
#define DM_C     64
#define H_C      8
#define DH_C     8
#define DFF_C    256
#define PRED_C   96
#define UTOP_C   35             // u = U_part = 35

typedef __attribute__((ext_vector_type(16))) __bf16 v16bf;
typedef __attribute__((ext_vector_type(8)))  __bf16 bf16x8;
typedef __attribute__((ext_vector_type(8)))  float  v8f;

// ---------------- helpers ----------------
__device__ __forceinline__ __bf16 f2bf(float f) {
    unsigned u = __float_as_uint(f);
    unsigned r = (u + 0x7FFFu + ((u >> 16) & 1u)) >> 16;
    unsigned short s = (unsigned short)r;
    return __builtin_bit_cast(__bf16, s);
}

__device__ __forceinline__ unsigned hashu(unsigned x) {
    x ^= x >> 16; x *= 0x7feb352dU;
    x ^= x >> 15; x *= 0x846ca68bU;
    x ^= x >> 16; return x;
}

__device__ __forceinline__ float wave_sum(float v) {
    #pragma unroll
    for (int o = 16; o > 0; o >>= 1) v += __shfl_xor(v, o, 32);
    return v;
}

// ---------------- generic f32 -> bf16 convert ----------------
__global__ void cvt_kernel(const float* __restrict__ src, __bf16* __restrict__ dst, size_t n) {
    size_t i = (size_t)blockIdx.x * blockDim.x + threadIdx.x;
    if (i < n) dst[i] = f2bf(src[i]);
}

__global__ void zero_kernel(float* __restrict__ p, size_t n) {
    size_t i = (size_t)blockIdx.x * blockDim.x + threadIdx.x;
    if (i < n) p[i] = 0.0f;
}

// ---------------- generic WMMA GEMM on bf16 operands ----------------
// C[M,N] = A[M,K](bf16) @ W[N,K](bf16)^T + bias (+residual f32) (+GELU)
// one wave per 16 x (16*NT) strip; grid = (N/(16*NT), M/16, splitK)
// split-K: each z-chunk covers K range [z*KCHUNK, ...); accumulate via atomicAdd.
#define KCHUNK_C 4096
template <int NT>
__global__ void gemm_wmma_kernel(const __bf16* __restrict__ A, int lda,
                                 const __bf16* __restrict__ W, int ldw,
                                 const float* __restrict__ bias,
                                 const float* __restrict__ residual, int ldr,
                                 float* __restrict__ Cf, __bf16* __restrict__ Cb, int ldc,
                                 int K, int act) {
    const int tn0 = blockIdx.x * NT;
    const int tm  = blockIdx.y;
    const int lane = threadIdx.x & 31;
    const int row  = lane & 15;               // A row within tile / C col
    const int kbA  = (lane >> 4) << 4;        // 0 or 16
    const int kbB  = (lane >> 4) << 4;        // 0 or 16

    int kbeg = 0, kend = K;
    if (gridDim.z > 1) {
        kbeg = blockIdx.z * KCHUNK_C;
        kend = min(K, kbeg + KCHUNK_C);
    }

    const __bf16* Ap = A + ((size_t)tm * 16 + row) * lda;
    const __bf16* Wp[NT];
    #pragma unroll
    for (int t = 0; t < NT; ++t)
        Wp[t] = W + ((size_t)(tn0 + t) * 16 + row) * ldw + kbB;

    v8f zero = {};
    v8f acc[NT];
    #pragma unroll
    for (int t = 0; t < NT; ++t) acc[t] = zero;

    #pragma unroll 2
    for (int k0 = kbeg; k0 < kend; k0 += 32) {
        // A fragment: two contiguous 16B chunks, merged
        bf16x8 alo = *(const bf16x8*)(Ap + k0 + kbA);
        bf16x8 ahi = *(const bf16x8*)(Ap + k0 + 8 + kbA);
        v16bf a = __builtin_shufflevector(alo, ahi,
                      0, 1, 2, 3, 4, 5, 6, 7, 8, 9, 10, 11, 12, 13, 14, 15);
        __builtin_prefetch(Ap + k0 + 512, 0, 1);
        #pragma unroll
        for (int t = 0; t < NT; ++t) {
            v16bf b = *(const v16bf*)(Wp[t] + k0);   // 16 contiguous bf16
            acc[t] = __builtin_amdgcn_wmma_f32_16x16x32_bf16(
                         false, a, false, b, (short)0, acc[t], false, false);
        }
    }

    const int col = lane & 15;
    const int rb  = (lane >> 4) * 8;          // M offset 0 or 8
    const bool first = (kbeg == 0);
    #pragma unroll
    for (int t = 0; t < NT; ++t) {
        int gn = (tn0 + t) * 16 + col;
        float bv = (bias && first) ? bias[gn] : 0.0f;
        #pragma unroll
        for (int r = 0; r < 8; ++r) {
            int gm = tm * 16 + rb + r;
            float v = acc[t][r] + bv;
            if (residual && first) v += residual[(size_t)gm * ldr + gn];
            if (act == 1) v = 0.5f * v * (1.0f + erff(v * 0.70710678118f));
            if (gridDim.z > 1) {
                atomicAdd(&Cf[(size_t)gm * ldc + gn], v);
            } else {
                if (Cf) Cf[(size_t)gm * ldc + gn] = v;
                if (Cb) Cb[(size_t)gm * ldc + gn] = f2bf(v);
            }
        }
    }
}

// ---------------- patch gather: x_enc (B,SL,M) -> patches bf16 (ROWS, 32) ----
__global__ void patch_kernel(const float* __restrict__ x_enc, __bf16* __restrict__ patches) {
    size_t i = (size_t)blockIdx.x * blockDim.x + threadIdx.x;
    if (i >= (size_t)ROWS_C * 32) return;
    int col = (int)(i & 31);
    size_t r = i >> 5;
    int pi = (int)(r % NP_C);
    int bm = (int)(r / NP_C);
    int b = bm >> 3, m = bm & 7;
    float val = 0.0f;
    if (col < P_C) {
        int t = pi * STRIDE_C + col;
        val = x_enc[((size_t)b * SL_C + t) * 8 + m];
    }
    patches[i] = f2bf(val);
}

// ---------------- pad in_W (64,16) -> bf16 (64,32) ----------------
__global__ void padw_kernel(const float* __restrict__ w, __bf16* __restrict__ wp) {
    int i = blockIdx.x * blockDim.x + threadIdx.x;   // 64*32
    if (i >= DM_C * 32) return;
    int n = i >> 5, k = i & 31;
    wp[i] = f2bf((k < P_C) ? w[n * P_C + k] : 0.0f);
}

// ---------------- sparsity measure (ProbSparse sampling) ----------------
__global__ void sparsity_kernel(const float* __restrict__ q, const float* __restrict__ k,
                                float* __restrict__ spars, int layer) {
    int bm = blockIdx.x, h = blockIdx.y;
    for (int l = threadIdx.x; l < NP_C; l += blockDim.x) {
        float qv[8];
        const float* qp = q + ((size_t)(bm * NP_C + l)) * DM_C + h * DH_C;
        #pragma unroll
        for (int j = 0; j < 8; ++j) qv[j] = qp[j];
        float mx = -1e30f, sm = 0.0f;
        for (int u = 0; u < UTOP_C; ++u) {
            unsigned hsh = hashu((unsigned)(layer * 0x9E3779B9u) ^
                                 (unsigned)(l * 911u + u * 2654435761u));
            int idx = (int)(hsh % NP_C);
            const float* kp = k + ((size_t)(bm * NP_C + idx)) * DM_C + h * DH_C;
            float d = 0.0f;
            #pragma unroll
            for (int j = 0; j < 8; ++j) d += qv[j] * kp[j];
            mx = fmaxf(mx, d);
            sm += d;
        }
        spars[((size_t)(bm * H_C + h)) * NP_C + l] = mx - sm * (1.0f / (float)NP_C);
    }
}

// ---------------- top-k (u=35) per (bm,h): one wave per block ----------------
__global__ void topk_kernel(const float* __restrict__ spars, int* __restrict__ top_idx) {
    int bh = blockIdx.x;                  // bm*8+h
    __shared__ float s[NP_C];
    int lane = threadIdx.x;
    for (int l = lane; l < NP_C; l += 32) s[l] = spars[(size_t)bh * NP_C + l];
    __syncthreads();
    for (int u = 0; u < UTOP_C; ++u) {
        float bv = -1e30f; int bi = NP_C;
        for (int l = lane; l < NP_C; l += 32)
            if (s[l] > bv) { bv = s[l]; bi = l; }
        #pragma unroll
        for (int o = 16; o > 0; o >>= 1) {
            float ov = __shfl_xor(bv, o, 32);
            int   oi = __shfl_xor(bi, o, 32);
            if (ov > bv || (ov == bv && oi < bi)) { bv = ov; bi = oi; }
        }
        if (lane == 0) { top_idx[bh * UTOP_C + u] = bi; s[bi] = -1e30f; }
        __syncthreads();
    }
}

// ---------------- ctx init: broadcast mean_l v over all patches (bf16 out) ----
__global__ void ctx_init_kernel(const float* __restrict__ v, __bf16* __restrict__ ctx) {
    int bm = blockIdx.x;
    __shared__ float part[4][DM_C];
    __shared__ float mean[DM_C];
    int t = threadIdx.x;
    int c = t & 63, g = t >> 6;
    float sum = 0.0f;
    for (int l = g; l < NP_C; l += 4)
        sum += v[((size_t)(bm * NP_C + l)) * DM_C + c];
    part[g][c] = sum;
    __syncthreads();
    if (g == 0)
        mean[c] = (part[0][c] + part[1][c] + part[2][c] + part[3][c]) * (1.0f / (float)NP_C);
    __syncthreads();
    size_t base = (size_t)bm * NP_C * DM_C;
    for (int i = t; i < NP_C * DM_C; i += 256)
        ctx[base + i] = f2bf(mean[i & 63]);
}

// ---------------- sparse attention on top-u queries (online softmax) ----------
__global__ void attn_kernel(const float* __restrict__ q, const float* __restrict__ k,
                            const float* __restrict__ v, const int* __restrict__ top_idx,
                            __bf16* __restrict__ ctx) {
    int bm = blockIdx.x, h = blockIdx.y;
    int wave = threadIdx.x >> 5, lane = threadIdx.x & 31;
    const float scale = 0.3535533906f;    // rsqrt(8)
    for (int u = wave; u < UTOP_C; u += 8) {
        int qi = top_idx[(bm * H_C + h) * UTOP_C + u];
        float qv[8];
        const float* qp = q + ((size_t)(bm * NP_C + qi)) * DM_C + h * DH_C;
        #pragma unroll
        for (int j = 0; j < 8; ++j) qv[j] = qp[j];
        float m = -1e30f, s = 0.0f, acc[8];
        #pragma unroll
        for (int j = 0; j < 8; ++j) acc[j] = 0.0f;
        for (int kk = lane; kk < NP_C; kk += 32) {
            const float* kp = k + ((size_t)(bm * NP_C + kk)) * DM_C + h * DH_C;
            float d = 0.0f;
            #pragma unroll
            for (int j = 0; j < 8; ++j) d += qv[j] * kp[j];
            d *= scale;
            float mn = fmaxf(m, d);
            float c1 = __expf(m - mn), c2 = __expf(d - mn);
            const float* vp = v + ((size_t)(bm * NP_C + kk)) * DM_C + h * DH_C;
            s = s * c1 + c2;
            #pragma unroll
            for (int j = 0; j < 8; ++j) acc[j] = acc[j] * c1 + c2 * vp[j];
            m = mn;
        }
        #pragma unroll
        for (int o = 16; o > 0; o >>= 1) {
            float mo = __shfl_xor(m, o, 32);
            float so = __shfl_xor(s, o, 32);
            float ao[8];
            #pragma unroll
            for (int j = 0; j < 8; ++j) ao[j] = __shfl_xor(acc[j], o, 32);
            float mn = fmaxf(m, mo);
            float c1 = __expf(m - mn), c2 = __expf(mo - mn);
            s = s * c1 + so * c2;
            #pragma unroll
            for (int j = 0; j < 8; ++j) acc[j] = acc[j] * c1 + ao[j] * c2;
            m = mn;
        }
        if (lane < 8)
            ctx[((size_t)(bm * NP_C + qi)) * DM_C + h * DH_C + lane] = f2bf(acc[lane] / s);
    }
}

// ---------------- LayerNorm over last dim (64), in place + bf16 shadow ----
__global__ void ln_kernel(float* __restrict__ x, const float* __restrict__ g,
                          const float* __restrict__ b, __bf16* __restrict__ xb, int rows) {
    int wave = (int)(((size_t)blockIdx.x * blockDim.x + threadIdx.x) >> 5);
    int lane = threadIdx.x & 31;
    if (wave >= rows) return;
    float* row = x + (size_t)wave * DM_C;
    float v0 = row[lane], v1 = row[lane + 32];
    float mu = wave_sum(v0 + v1) * (1.0f / 64.0f);
    float d0 = v0 - mu, d1 = v1 - mu;
    float var = wave_sum(d0 * d0 + d1 * d1) * (1.0f / 64.0f);
    float inv = rsqrtf(var + 1e-5f);
    float o0 = d0 * inv * g[lane]      + b[lane];
    float o1 = d1 * inv * g[lane + 32] + b[lane + 32];
    row[lane]      = o0;
    row[lane + 32] = o1;
    if (xb) {
        __bf16* rb = xb + (size_t)wave * DM_C;
        rb[lane]      = f2bf(o0);
        rb[lane + 32] = f2bf(o1);
    }
}

// ---------------- output transpose: (B*M, PRED) -> (B, PRED, M) ----------------
__global__ void out_tr_kernel(const float* __restrict__ C, float* __restrict__ out) {
    int i = blockIdx.x * blockDim.x + threadIdx.x;      // 8*96*8
    if (i >= 8 * PRED_C * 8) return;
    int m = i & 7;
    int p = (i >> 3) % PRED_C;
    int b = i / (PRED_C * 8);
    out[i] = C[((size_t)(b * 8 + m)) * PRED_C + p];
}

// ---------------- host-side launch ----------------
extern "C" void kernel_launch(void* const* d_in, const int* in_sizes, int n_in,
                              void* d_out, int out_size, void* d_ws, size_t ws_size,
                              hipStream_t stream) {
    (void)in_sizes; (void)n_in; (void)out_size; (void)ws_size;
    const float* x_enc = (const float*)d_in[0];
    const float* in_W  = (const float*)d_in[4];
    const float* in_b  = (const float*)d_in[5];
    const float* Wq    = (const float*)d_in[6];
    const float* bq    = (const float*)d_in[7];
    const float* Wk    = (const float*)d_in[8];
    const float* bk    = (const float*)d_in[9];
    const float* Wv    = (const float*)d_in[10];
    const float* bv    = (const float*)d_in[11];
    const float* Wo    = (const float*)d_in[12];
    const float* bo    = (const float*)d_in[13];
    const float* c1_W  = (const float*)d_in[14];
    const float* c1_b  = (const float*)d_in[15];
    const float* c2_W  = (const float*)d_in[16];
    const float* c2_b  = (const float*)d_in[17];
    const float* ln1_g = (const float*)d_in[18];
    const float* ln1_b = (const float*)d_in[19];
    const float* ln2_g = (const float*)d_in[20];
    const float* ln2_b = (const float*)d_in[21];
    const float* lnf_g = (const float*)d_in[22];
    const float* lnf_b = (const float*)d_in[23];
    const float* out_W = (const float*)d_in[24];
    const float* out_b = (const float*)d_in[25];
    float* out = (float*)d_out;

    // workspace carve-up
    char* base = (char*)d_ws;
    size_t off = 0;
    auto allocB = [&](size_t bytes) {
        void* p = base + off;
        off += (bytes + 255) & ~(size_t)255;
        return p;
    };
    auto allocF  = [&](size_t n) { return (float*)allocB(n * 4); };
    auto allocH  = [&](size_t n) { return (__bf16*)allocB(n * 2); };

    // f32 activations
    float* zbuf  = allocF((size_t)ROWS_C * DM_C);
    float* qbuf  = allocF((size_t)ROWS_C * DM_C);
    float* kbuf  = allocF((size_t)ROWS_C * DM_C);
    float* vbuf  = allocF((size_t)ROWS_C * DM_C);
    float* x1buf = allocF((size_t)ROWS_C * DM_C);
    float* spars = allocF((size_t)BM_C * H_C * NP_C);
    float* chead = allocF((size_t)BM_C * PRED_C);
    int*   top_idx = (int*)allocB((size_t)BM_C * H_C * UTOP_C * 4);
    // bf16 shadows / operands
    __bf16* patches_b = allocH((size_t)ROWS_C * 32);
    __bf16* inWpad_b  = allocH((size_t)DM_C * 32);
    __bf16* z_b   = allocH((size_t)ROWS_C * DM_C);
    __bf16* ctx_b = allocH((size_t)ROWS_C * DM_C);
    __bf16* x1_b  = allocH((size_t)ROWS_C * DM_C);
    __bf16* y_b   = allocH((size_t)ROWS_C * DFF_C);
    __bf16* wq_b  = allocH((size_t)2 * DM_C * DM_C);
    __bf16* wk_b  = allocH((size_t)2 * DM_C * DM_C);
    __bf16* wv_b  = allocH((size_t)2 * DM_C * DM_C);
    __bf16* wo_b  = allocH((size_t)2 * DM_C * DM_C);
    __bf16* c1w_b = allocH((size_t)2 * DFF_C * DM_C);
    __bf16* c2w_b = allocH((size_t)2 * DM_C * DFF_C);
    __bf16* outW_b = allocH((size_t)PRED_C * ROWS_C);

    const int TM = ROWS_C / 16;          // 2044
    auto cgrid = [](size_t n) { return dim3((unsigned)((n + 255) / 256)); };

    // one-time weight conversions to bf16
    cvt_kernel<<<cgrid(2 * DM_C * DM_C), 256, 0, stream>>>(Wq, wq_b, (size_t)2 * DM_C * DM_C);
    cvt_kernel<<<cgrid(2 * DM_C * DM_C), 256, 0, stream>>>(Wk, wk_b, (size_t)2 * DM_C * DM_C);
    cvt_kernel<<<cgrid(2 * DM_C * DM_C), 256, 0, stream>>>(Wv, wv_b, (size_t)2 * DM_C * DM_C);
    cvt_kernel<<<cgrid(2 * DM_C * DM_C), 256, 0, stream>>>(Wo, wo_b, (size_t)2 * DM_C * DM_C);
    cvt_kernel<<<cgrid(2 * DFF_C * DM_C), 256, 0, stream>>>(c1_W, c1w_b, (size_t)2 * DFF_C * DM_C);
    cvt_kernel<<<cgrid(2 * DM_C * DFF_C), 256, 0, stream>>>(c2_W, c2w_b, (size_t)2 * DM_C * DFF_C);
    cvt_kernel<<<cgrid((size_t)PRED_C * ROWS_C), 256, 0, stream>>>(out_W, outW_b, (size_t)PRED_C * ROWS_C);

    // patch gather + padded patch-embedding weight
    patch_kernel<<<cgrid((size_t)ROWS_C * 32), 256, 0, stream>>>(x_enc, patches_b);
    padw_kernel<<<cgrid(DM_C * 32), 256, 0, stream>>>(in_W, inWpad_b);

    // patch embedding: z = patches @ in_W^T + in_b  (f32 + bf16 shadow)
    gemm_wmma_kernel<4><<<dim3(1, TM, 1), 32, 0, stream>>>(
        patches_b, 32, inWpad_b, 32, in_b, nullptr, 0, zbuf, z_b, DM_C, 32, 0);

    for (int e = 0; e < 2; ++e) {
        const __bf16* wq_e = wq_b + (size_t)e * DM_C * DM_C;
        const __bf16* wk_e = wk_b + (size_t)e * DM_C * DM_C;
        const __bf16* wv_e = wv_b + (size_t)e * DM_C * DM_C;
        const __bf16* wo_e = wo_b + (size_t)e * DM_C * DM_C;
        const __bf16* c1w_e = c1w_b + (size_t)e * DFF_C * DM_C;
        const __bf16* c2w_e = c2w_b + (size_t)e * DM_C * DFF_C;

        // QKV projections (f32 outputs for the VALU attention part)
        gemm_wmma_kernel<4><<<dim3(1, TM, 1), 32, 0, stream>>>(
            z_b, DM_C, wq_e, DM_C, bq + e * DM_C, nullptr, 0, qbuf, nullptr, DM_C, DM_C, 0);
        gemm_wmma_kernel<4><<<dim3(1, TM, 1), 32, 0, stream>>>(
            z_b, DM_C, wk_e, DM_C, bk + e * DM_C, nullptr, 0, kbuf, nullptr, DM_C, DM_C, 0);
        gemm_wmma_kernel<4><<<dim3(1, TM, 1), 32, 0, stream>>>(
            z_b, DM_C, wv_e, DM_C, bv + e * DM_C, nullptr, 0, vbuf, nullptr, DM_C, DM_C, 0);

        // ProbSparse attention
        sparsity_kernel<<<dim3(BM_C, H_C), 256, 0, stream>>>(qbuf, kbuf, spars, e);
        topk_kernel<<<dim3(BM_C * H_C), 32, 0, stream>>>(spars, top_idx);
        ctx_init_kernel<<<dim3(BM_C), 256, 0, stream>>>(vbuf, ctx_b);
        attn_kernel<<<dim3(BM_C, H_C), 256, 0, stream>>>(qbuf, kbuf, vbuf, top_idx, ctx_b);

        // output projection + residual, then LN1 (writes f32 + bf16 shadow)
        gemm_wmma_kernel<4><<<dim3(1, TM, 1), 32, 0, stream>>>(
            ctx_b, DM_C, wo_e, DM_C, bo + e * DM_C, zbuf, DM_C, x1buf, nullptr, DM_C, DM_C, 0);
        ln_kernel<<<dim3(ROWS_C / 8), 256, 0, stream>>>(
            x1buf, ln1_g + e * DM_C, ln1_b + e * DM_C, x1_b, ROWS_C);

        // FFN: GELU(x1 @ c1^T) @ c2^T + residual, then LN2
        gemm_wmma_kernel<4><<<dim3(DFF_C / 64, TM, 1), 32, 0, stream>>>(
            x1_b, DM_C, c1w_e, DM_C, c1_b + e * DFF_C, nullptr, 0, nullptr, y_b, DFF_C, DM_C, 1);
        gemm_wmma_kernel<4><<<dim3(1, TM, 1), 32, 0, stream>>>(
            y_b, DFF_C, c2w_e, DFF_C, c2_b + e * DM_C, x1buf, DM_C, zbuf, nullptr, DM_C, DFF_C, 0);
        ln_kernel<<<dim3(ROWS_C / 8), 256, 0, stream>>>(
            zbuf, ln2_g + e * DM_C, ln2_b + e * DM_C, z_b, ROWS_C);
    }

    // final LN (z_b becomes the head's A operand, viewed as (64, 32704))
    ln_kernel<<<dim3(ROWS_C / 8), 256, 0, stream>>>(zbuf, lnf_g, lnf_b, z_b, ROWS_C);

    // head: (64, 32704) @ (96, 32704)^T with split-K over gridDim.z
    zero_kernel<<<cgrid((size_t)BM_C * PRED_C), 256, 0, stream>>>(chead, (size_t)BM_C * PRED_C);
    gemm_wmma_kernel<3><<<dim3(2, BM_C / 16, 8), 32, 0, stream>>>(
        z_b, ROWS_C, outW_b, ROWS_C, out_b, nullptr, 0, chead, nullptr, PRED_C, ROWS_C, 0);

    // (B*M, PRED) -> (B, PRED, M)
    out_tr_kernel<<<cgrid(8 * PRED_C * 8), 256, 0, stream>>>(chead, out);
}
